// TEM_7344394076339
// MI455X (gfx1250) — matrix operand order repaired and checked by reading, not compiled, not run on hardware
//
#include <hip/hip_runtime.h>
#include <hip/hip_bf16.h>
#include <math.h>

typedef _Float16 h8   __attribute__((ext_vector_type(8)));
typedef _Float16 v16h __attribute__((ext_vector_type(16)));
typedef float    v8f  __attribute__((ext_vector_type(8)));

#define NB   4
#define CIN  256
#define CM   128
#define HH   128
#define WWD  128
#define HW   (HH*WWD)      // 16384
#define PH   130           // padded height (1-px halo)
#define PW   130           // padded width
#define LQ   128
#define PP   (120*120)     // 14400
#define BN_SMUL 0.9999950000374997f   // 1/sqrt(1+1e-5)

__device__ __forceinline__ int imin(int a, int b){ return a < b ? a : b; }
__device__ __forceinline__ int imax(int a, int b){ return a > b ? a : b; }

// ---------------- WMMA fragment loaders ----------------
// A (16x32 f16): lane = row m0+lane%16; K = k0+(lane/16)*16+e
__device__ __forceinline__ v16h ld_a_frag(const _Float16* A, int lda, int m0, int k0, int lane){
    int lm = lane & 15, lh = lane >> 4;
    const _Float16* p = A + (size_t)(m0 + lm) * lda + k0 + lh * 16;
    union U { v16h v; struct { h8 lo, hi; } s; } u;
    u.s.lo = *(const h8*)p;
    u.s.hi = *(const h8*)(p + 8);
    return u.v;
}
// B (32x16 f16) from Bt row-major [N,K]: lane = col n0+lane%16; K = k0+(lane/16)*16+e
__device__ __forceinline__ v16h ld_b_frag(const _Float16* Bt, int ldb, int n0, int k0, int lane){
    int lm = lane & 15, lh = lane >> 4;
    return *(const v16h*)(Bt + (size_t)(n0 + lm) * ldb + k0 + lh * 16);
}

// ---------------- generic WMMA GEMM, NT n-tiles per wave ----------------
// C[M,Nc] = A[M,K] * Bt[N,K]^T ; fused scale/bias per-m, activation, f32/f16 outputs.
template<int NT>
__global__ void qco_gemm16(const _Float16* __restrict__ A, int lda,
                           const _Float16* __restrict__ Bt, int ldb,
                           float* C, long ldc, int ctrans,
                           _Float16* C16, long ldc16, int c16trans,
                           const float* gamma, const float* beta, float smul, int act,
                           int M, int Nc, int K)
{
    int lane = threadIdx.x & 31;
    int wid  = blockIdx.x * (blockDim.x >> 5) + (threadIdx.x >> 5);
    int mtiles = M >> 4;
    int ngrp   = Nc / (16 * NT);
    int tiles  = mtiles * ngrp;
    if (wid >= tiles) return;
    int m0 = (wid % mtiles) << 4;
    int n0 = (wid / mtiles) * (16 * NT);

    v8f acc[NT];
    #pragma unroll
    for (int j = 0; j < NT; ++j) acc[j] = (v8f){};

    for (int k0 = 0; k0 < K; k0 += 32) {
        v16h a = ld_a_frag(A, lda, m0, k0, lane);     // reused across NT tiles
        #pragma unroll
        for (int j = 0; j < NT; ++j) {
            v16h b = ld_b_frag(Bt, ldb, n0 + 16 * j, k0, lane);
            acc[j] = __builtin_amdgcn_wmma_f32_16x16x32_f16(false, a, false, b, (short)0, acc[j], false, false);
        }
    }

    int lm = lane & 15, lh = lane >> 4;
    #pragma unroll
    for (int j = 0; j < NT; ++j) {
        int ncol = n0 + 16 * j + lm;
        float vals[8];
        #pragma unroll
        for (int v = 0; v < 8; ++v) {
            int m = m0 + 8 * lh + v;
            float x = acc[j][v];
            if (gamma) x = x * (gamma[m] * smul) + beta[m];
            if (act == 1)      x = fmaxf(x, 0.f);
            else if (act == 2) x = x > 0.f ? x : 0.01f * x;
            vals[v] = x;
        }
        if (C) {
            if (ctrans) {
                float* p = C + (size_t)ncol * ldc + m0 + 8 * lh;   // contiguous 8 floats
                #pragma unroll
                for (int v = 0; v < 8; ++v) p[v] = vals[v];
            } else {
                #pragma unroll
                for (int v = 0; v < 8; ++v) C[(size_t)(m0 + 8*lh + v) * ldc + ncol] = vals[v];
            }
        }
        if (C16) {
            if (c16trans) {
                h8 hv;
                #pragma unroll
                for (int v = 0; v < 8; ++v) hv[v] = (_Float16)vals[v];
                *(h8*)(C16 + (size_t)ncol * ldc16 + m0 + 8 * lh) = hv;   // contiguous 16B
            } else {
                #pragma unroll
                for (int v = 0; v < 8; ++v) C16[(size_t)(m0 + 8*lh + v) * ldc16 + ncol] = (_Float16)vals[v];
            }
        }
    }
}

// ---------------- conv3x3 stem: 9 shifted GEMMs + BN + LeakyReLU ----------------
// x16p is NHWC f16 with a 1-pixel zero halo (PHxPW rows) -> all loads unconditional.
// Each wave: 16 oc x 64 pixels (4 accumulator chains, A-fragment reuse).
__global__ void qco_conv3_wmma(const _Float16* __restrict__ x16p, const _Float16* __restrict__ w3p,
                               const float* __restrict__ g, const float* __restrict__ bb,
                               _Float16* __restrict__ h16)
{
    int lane = threadIdx.x & 31;
    int wid  = blockIdx.x * 8 + (threadIdx.x >> 5);
    int mt = wid & 15;          // 16 oc tiles
    int nt = wid >> 4;          // NB*HW/64 = 1024 pixel groups
    int oc0 = mt << 4;
    int pb  = nt << 6;          // 64-pixel base, stays within one row (W=128)
    int n   = pb >> 14;
    int p0  = pb & (HW - 1);
    int y   = p0 >> 7;
    int lm = lane & 15, lh = lane >> 4;
    int xb = (p0 & 127) + lm;   // unpadded x of this lane's column in tile 0

    v8f acc[4];
    #pragma unroll
    for (int j = 0; j < 4; ++j) acc[j] = (v8f){};

    // lane's pixel row pointer in padded image, tile j at +16*j pixels
    const _Float16* prow = x16p + ((size_t)n * PH * PW + (size_t)(y + 1) * PW + (xb + 1)) * CIN;

    #pragma unroll
    for (int dy = -1; dy <= 1; ++dy) {
        #pragma unroll
        for (int dx = -1; dx <= 1; ++dx) {
            const _Float16* abase = w3p + (size_t)((dy + 1) * 3 + (dx + 1)) * CIN * CIN;
            const _Float16* b0 = prow + ((long)dy * PW + dx) * CIN;
            for (int c0 = 0; c0 < CIN; c0 += 32) {
                v16h a = ld_a_frag(abase, CIN, oc0, c0, lane);   // reused for 4 tiles
                #pragma unroll
                for (int j = 0; j < 4; ++j) {
                    v16h b = *(const v16h*)(b0 + (size_t)(16 * j) * CIN + c0 + lh * 16);
                    acc[j] = __builtin_amdgcn_wmma_f32_16x16x32_f16(false, a, false, b, (short)0, acc[j], false, false);
                }
            }
        }
    }
    #pragma unroll
    for (int j = 0; j < 4; ++j) {
        h8 hv;
        #pragma unroll
        for (int v = 0; v < 8; ++v) {
            int oc = oc0 + 8 * lh + v;
            float val = acc[j][v] * (g[oc] * BN_SMUL) + bb[oc];
            val = val > 0.f ? val : 0.01f * val;
            hv[v] = (_Float16)val;
        }
        *(h8*)(h16 + ((size_t)(pb + 16 * j + lm)) * CIN + oc0 + 8 * lh) = hv;  // NHWC f16
    }
}

// ---------------- packing / conversion ----------------
__global__ void qco_zero16(_Float16* __restrict__ p, long n){
    long i = (long)blockIdx.x * blockDim.x + threadIdx.x;
    if (i < n) p[i] = (_Float16)0.f;
}
__global__ void qco_pack_x(const float* __restrict__ x, _Float16* __restrict__ x16p, long total){
    long i = (long)blockIdx.x * blockDim.x + threadIdx.x;
    if (i >= total) return;
    long p = i & (HW - 1);
    long c = (i >> 14) & (CIN - 1);
    long n = i >> 22;
    long yy = (p >> 7) + 1, xx = (p & 127) + 1;
    x16p[((n * PH + yy) * PW + xx) * CIN + c] = (_Float16)x[i];
}
__global__ void qco_pack_w3(const float* __restrict__ w, _Float16* __restrict__ w3p, int total){
    int i = blockIdx.x * blockDim.x + threadIdx.x;
    if (i >= total) return;
    int oc = i / (CIN * 9);
    int r  = i % (CIN * 9);
    int c  = r / 9;
    int t  = r % 9;
    w3p[(size_t)t * CIN * CIN + (size_t)oc * CIN + c] = (_Float16)w[i];
}
__global__ void qco_cvt16(const float* __restrict__ s, _Float16* __restrict__ d, int n){
    int i = blockIdx.x * blockDim.x + threadIdx.x;
    if (i < n) d[i] = (_Float16)s[i];
}
__global__ void qco_zero(float* p, int n){
    int i = blockIdx.x * blockDim.x + threadIdx.x;
    if (i < n) p[i] = 0.f;
}

// ---------------- statistics kernels ----------------
__global__ void qco_reduce_xav(const float* __restrict__ xbr, float* __restrict__ xav, int weighted){
    int n = blockIdx.x >> 7;
    int c = blockIdx.x & 127;
    __shared__ float sm[256];
    float s = 0.f;
    for (int p = threadIdx.x; p < HW; p += 256) {
        float w = 1.f;
        if (weighted) {
            int y = p >> 7, x = p & 127;
            int cx = imin(x, 119) - imax(0, x - 8) + 1;
            int cy = imin(y, 119) - imax(0, y - 8) + 1;
            w = (float)(cx * cy);
        }
        s += xbr[((size_t)n * HW + p) * CM + c] * w;
    }
    sm[threadIdx.x] = s; __syncthreads();
    for (int st = 128; st > 0; st >>= 1) {
        if (threadIdx.x < st) sm[threadIdx.x] += sm[threadIdx.x + st];
        __syncthreads();
    }
    if (threadIdx.x == 0)
        xav[n * CM + c] = sm[0] * (weighted ? (1.f / 1166400.f) : (1.f / (float)HW));
}
__global__ void qco_xavn(const float* __restrict__ xav, float* __restrict__ xavn){
    int n = blockIdx.x;
    __shared__ float sm[128];
    float v = xav[n * CM + threadIdx.x];
    sm[threadIdx.x] = v * v; __syncthreads();
    for (int st = 64; st > 0; st >>= 1) {
        if (threadIdx.x < st) sm[threadIdx.x] += sm[threadIdx.x + st];
        __syncthreads();
    }
    float nrm = fmaxf(sqrtf(sm[0]), 1e-12f);
    xavn[n * CM + threadIdx.x] = v / nrm;
}
__global__ void qco_cs_c(const float* __restrict__ xbr, const float* __restrict__ xavn, float* __restrict__ cs){
    int gp = blockIdx.x * blockDim.x + threadIdx.x;
    if (gp >= NB * HW) return;
    int n = gp >> 14;
    const float* row = xbr + (size_t)gp * CM;
    const float* av  = xavn + n * CM;
    float dot = 0.f, nn = 0.f;
    for (int c = 0; c < CM; ++c) { float v = row[c]; dot += v * av[c]; nn += v * v; }
    cs[gp] = dot / fmaxf(sqrtf(nn), 1e-12f);
}
__global__ void qco_xsum(const float* __restrict__ xbr, float* __restrict__ xsum){
    int gp = blockIdx.x * blockDim.x + threadIdx.x;
    if (gp >= NB * HW) return;
    const float* row = xbr + (size_t)gp * CM;
    float s = 0.f;
    for (int c = 0; c < CM; ++c) s += row[c];
    xsum[gp] = s;
}
__global__ void qco_pavg(const float* __restrict__ xsum, float* __restrict__ pavg){
    int id = blockIdx.x * blockDim.x + threadIdx.x;
    if (id >= NB * PP) return;
    int n = id / PP, r = id % PP;
    int sy = r / 120, sx = r % 120;
    float s = 0.f;
    for (int dy = 0; dy < 9; ++dy)
        for (int dx = 0; dx < 9; ++dx)
            s += xsum[(size_t)n * HW + (sy + dy) * WWD + sx + dx];
    pavg[id] = s * (1.f / 81.f);
}
__global__ void qco_normj(const float* __restrict__ xsum, float* __restrict__ normj){
    int n = blockIdx.x / 81, j = blockIdx.x % 81;
    int dy = j / 9, dx = j % 9;
    __shared__ float sm[256];
    float s = 0.f;
    for (int i = threadIdx.x; i < PP; i += 256) {
        int yy = dy + i / 120, xx = dx + i % 120;
        float v = xsum[(size_t)n * HW + yy * WWD + xx];
        s += v * v;
    }
    sm[threadIdx.x] = s; __syncthreads();
    for (int st = 128; st > 0; st >>= 1) {
        if (threadIdx.x < st) sm[threadIdx.x] += sm[threadIdx.x + st];
        __syncthreads();
    }
    if (threadIdx.x == 0) normj[blockIdx.x] = fmaxf(sqrtf(sm[0]), 1e-12f);
}
__global__ void qco_navg(const float* __restrict__ pavg, float* __restrict__ navg){
    int n = blockIdx.x;
    __shared__ float sm[256];
    float s = 0.f;
    for (int i = threadIdx.x; i < PP; i += 256) { float v = pavg[(size_t)n * PP + i]; s += v * v; }
    sm[threadIdx.x] = s; __syncthreads();
    for (int st = 128; st > 0; st >>= 1) {
        if (threadIdx.x < st) sm[threadIdx.x] += sm[threadIdx.x + st];
        __syncthreads();
    }
    if (threadIdx.x == 0) navg[n] = fmaxf(sqrtf(sm[0]), 1e-12f);
}
__global__ void qco_folded(const float* __restrict__ xsum, const float* __restrict__ pavg,
                           const float* __restrict__ normj, const float* __restrict__ navg,
                           float* __restrict__ folded){
    int gp = blockIdx.x * blockDim.x + threadIdx.x;
    if (gp >= NB * HW) return;
    int n = gp >> 14, p = gp & (HW - 1);
    int y = p >> 7, x = p & 127;
    float s = 0.f;
    for (int j = 0; j < 81; ++j) {
        int dy = j / 9, dx = j % 9;
        int sy = y - dy, sx = x - dx;
        if ((unsigned)sy < 120u && (unsigned)sx < 120u)
            s += pavg[(size_t)n * PP + sy * 120 + sx] / normj[n * 81 + j];
    }
    folded[gp] = xsum[gp] / navg[n] * s;
}
__global__ void qco_cs_s(const float* __restrict__ folded, float* __restrict__ cs){
    int gp = blockIdx.x * blockDim.x + threadIdx.x;
    if (gp >= NB * HW) return;
    int n = gp >> 14, p = gp & (HW - 1);
    int y = p >> 7, x = p & 127;
    float m = -1e30f;
    for (int dy = -4; dy <= 4; ++dy)
        for (int dx = -4; dx <= 4; ++dx) {
            int yy = y + dy, xx = x + dx;
            float v = ((unsigned)yy < (unsigned)HH && (unsigned)xx < (unsigned)WWD)
                        ? folded[(size_t)n * HW + yy * WWD + xx] : 0.f;
            m = fmaxf(m, v);
        }
    cs[gp] = m;
}
__global__ void qco_minmax(const float* __restrict__ cs, float* __restrict__ cmm){
    int n = blockIdx.x;
    __shared__ float smn[256], smx[256];
    float mn = 1e30f, mx = -1e30f;
    for (int p = threadIdx.x; p < HW; p += 256) {
        float v = cs[(size_t)n * HW + p];
        mn = fminf(mn, v); mx = fmaxf(mx, v);
    }
    smn[threadIdx.x] = mn; smx[threadIdx.x] = mx; __syncthreads();
    for (int st = 128; st > 0; st >>= 1) {
        if (threadIdx.x < st) {
            smn[threadIdx.x] = fminf(smn[threadIdx.x], smn[threadIdx.x + st]);
            smx[threadIdx.x] = fmaxf(smx[threadIdx.x], smx[threadIdx.x + st]);
        }
        __syncthreads();
    }
    if (threadIdx.x == 0) { cmm[n * 2] = smn[0]; cmm[n * 2 + 1] = smx[0]; }
}
__global__ void qco_ql(const float* __restrict__ cmm, float* __restrict__ qlv){
    int id = blockIdx.x * blockDim.x + threadIdx.x;
    if (id >= NB * LQ) return;
    int n = id >> 7, l = id & 127;
    float mn = cmm[n * 2], mx = cmm[n * 2 + 1];
    qlv[id] = (float)(2 * l + 1) / (2.f * LQ) * (mx - mn) + mn;
}
__global__ void qco_quant_hist(const float* __restrict__ cs, const float* __restrict__ cmm,
                               _Float16* __restrict__ quant16, float* __restrict__ hist){
    int gp = blockIdx.x * blockDim.x + threadIdx.x;
    if (gp >= NB * HW) return;
    int n = gp >> 14;
    float c = cs[gp];
    float mn = cmm[n * 2], mx = cmm[n * 2 + 1];
    float range = mx - mn;
    float thr = 1.f - range / (float)LQ;
    _Float16* out = quant16 + (size_t)gp * LQ;
    for (int l = 0; l < LQ; ++l) {
        float ql = (float)(2 * l + 1) / (2.f * LQ) * range + mn;
        float q = 1.f - fabsf(ql - c);
        if (q > thr) {
            out[l] = (_Float16)q;
            atomicAdd(&hist[n * LQ + l], q);
        } else out[l] = (_Float16)0.f;
    }
}
__global__ void qco_hist_norm(float* __restrict__ hist){
    int n = blockIdx.x;
    __shared__ float sm[128];
    float v = hist[n * LQ + threadIdx.x];
    sm[threadIdx.x] = v; __syncthreads();
    for (int st = 64; st > 0; st >>= 1) {
        if (threadIdx.x < st) sm[threadIdx.x] += sm[threadIdx.x + st];
        __syncthreads();
    }
    hist[n * LQ + threadIdx.x] = v / sm[0];
}

// ---------------- sta head ----------------
__global__ void qco_sta1(const float* __restrict__ qlv, const float* __restrict__ hist,
                         const float* __restrict__ f1, float* __restrict__ s1){
    int id = blockIdx.x * blockDim.x + threadIdx.x;     // NB*64*128
    if (id >= NB * 64 * LQ) return;
    int n = id / (64 * LQ), r = id % (64 * LQ);
    int o = r / LQ, l = r % LQ;
    float v = f1[o * 2] * qlv[n * LQ + l] + f1[o * 2 + 1] * hist[n * LQ + l];
    v = v > 0.f ? v : 0.01f * v;
    s1[((size_t)n * LQ + l) * 64 + o] = v;
}
__global__ void qco_sta2(const float* __restrict__ f2, const float* __restrict__ s1, float* __restrict__ s2){
    int id = blockIdx.x * blockDim.x + threadIdx.x;     // NB*128*128
    if (id >= NB * CM * LQ) return;
    int n = id / (CM * LQ), r = id % (CM * LQ);
    int o = r / LQ, l = r % LQ;
    const float* srow = s1 + ((size_t)n * LQ + l) * 64;
    float v = 0.f;
    for (int c = 0; c < 64; ++c) v += f2[o * 64 + c] * srow[c];
    s2[((size_t)n * LQ + l) * CM + o] = fmaxf(v, 0.f);
}
__global__ void qco_sta3(const float* __restrict__ ow, const float* __restrict__ g, const float* __restrict__ b,
                         const float* __restrict__ s2, const float* __restrict__ xav,
                         _Float16* __restrict__ sta16){
    int id = blockIdx.x * blockDim.x + threadIdx.x;     // NB*128*128
    if (id >= NB * CM * LQ) return;
    int n = id / (CM * LQ), r = id % (CM * LQ);
    int o = r / LQ, l = r % LQ;
    const float* srow = s2 + ((size_t)n * LQ + l) * CM;
    const float* av = xav + n * CM;
    float v = 0.f;
    for (int c = 0; c < CM; ++c) v += ow[o * 256 + c] * srow[c];
    for (int c = 0; c < CM; ++c) v += ow[o * 256 + 128 + c] * av[c];
    v = v * (g[o] * BN_SMUL) + b[o];
    v = fmaxf(v, 0.f);
    sta16[((size_t)n * LQ + l) * CM + o] = (_Float16)v;   // [n][l][c] f16
}
__global__ void qco_softmax(const float* __restrict__ wmat, _Float16* __restrict__ wsoft){
    int id = blockIdx.x * blockDim.x + threadIdx.x;     // NB*128 rows
    if (id >= NB * LQ) return;
    const float* row = wmat + (size_t)id * LQ;
    float mx = -1e30f;
    for (int m = 0; m < LQ; ++m) mx = fmaxf(mx, row[m]);
    float s = 0.f;
    for (int m = 0; m < LQ; ++m) s += expf(row[m] - mx);
    float inv = 1.f / s;
    _Float16* o = wsoft + (size_t)id * LQ;
    for (int m = 0; m < LQ; ++m) o[m] = (_Float16)(expf(row[m] - mx) * inv);
}

// ---------------- host side ----------------
static inline void launch_gemm(hipStream_t st, const _Float16* A, int lda, const _Float16* Bt, int ldb,
                               float* C, long ldc, int ctrans, _Float16* C16, long ldc16, int c16t,
                               const float* g, const float* b, float smul, int act, int M, int Nc, int K)
{
    if (Nc % 64 == 0) {
        int tiles = (M / 16) * (Nc / 64);
        qco_gemm16<4><<<(tiles + 7) / 8, 256, 0, st>>>(A, lda, Bt, ldb, C, ldc, ctrans,
                                                       C16, ldc16, c16t, g, b, smul, act, M, Nc, K);
    } else if (Nc % 32 == 0) {
        int tiles = (M / 16) * (Nc / 32);
        qco_gemm16<2><<<(tiles + 7) / 8, 256, 0, st>>>(A, lda, Bt, ldb, C, ldc, ctrans,
                                                       C16, ldc16, c16t, g, b, smul, act, M, Nc, K);
    } else {
        int tiles = (M / 16) * (Nc / 16);
        qco_gemm16<1><<<(tiles + 7) / 8, 256, 0, st>>>(A, lda, Bt, ldb, C, ldc, ctrans,
                                                       C16, ldc16, c16t, g, b, smul, act, M, Nc, K);
    }
}

extern "C" void kernel_launch(void* const* d_in, const int* in_sizes, int n_in,
                              void* d_out, int out_size, void* d_ws, size_t ws_size,
                              hipStream_t stream)
{
    (void)in_sizes; (void)n_in; (void)out_size; (void)ws_size;
    char* base = (char*)d_ws;
    size_t off = 0;
    auto alloc = [&](size_t bytes) -> void* {
        off = (off + 255) & ~(size_t)255;
        void* p = base + off;
        off += bytes;
        return p;
    };
    _Float16* x16p  = (_Float16*)alloc((size_t)NB * PH * PW * CIN * 2);   // padded NHWC
    _Float16* h16   = (_Float16*)alloc((size_t)NB * HW * CIN * 2);
    _Float16* w3p   = (_Float16*)alloc((size_t)9 * CIN * CIN * 2);
    _Float16* w1_16 = (_Float16*)alloc((size_t)CM * CIN * 2);
    _Float16* kw16  = (_Float16*)alloc((size_t)CM * CM * 2);
    _Float16* qw16  = (_Float16*)alloc((size_t)CM * CM * 2);
    _Float16* vw16  = (_Float16*)alloc((size_t)CM * CM * 2);
    _Float16* ow16  = (_Float16*)alloc((size_t)256 * CM * 2);
    float*    xbr   = (float*)   alloc((size_t)NB * HW * CM * 4);
    _Float16* quant16=(_Float16*)alloc((size_t)NB * HW * LQ * 2);
    float*    cs    = (float*)   alloc((size_t)NB * HW * 4);
    float*    xsum  = (float*)   alloc((size_t)NB * HW * 4);
    float*    pavg  = (float*)   alloc((size_t)NB * PP * 4);
    float*    normj = (float*)   alloc((size_t)NB * 81 * 4);
    float*    navg  = (float*)   alloc((size_t)NB * 4);
    float*    folded= (float*)   alloc((size_t)NB * HW * 4);
    float*    cmm   = (float*)   alloc((size_t)NB * 2 * 4);
    float*    hist  = (float*)   alloc((size_t)NB * LQ * 4);
    float*    qlv   = (float*)   alloc((size_t)NB * LQ * 4);
    float*    xav   = (float*)   alloc((size_t)NB * CM * 4);
    float*    xavn  = (float*)   alloc((size_t)NB * CM * 4);
    float*    s1    = (float*)   alloc((size_t)NB * LQ * 64 * 4);
    float*    s2    = (float*)   alloc((size_t)NB * LQ * CM * 4);
    _Float16* sta16 = (_Float16*)alloc((size_t)NB * LQ * CM * 2);
    _Float16* kT16  = (_Float16*)alloc((size_t)NB * LQ * CM * 2);
    _Float16* qT16  = (_Float16*)alloc((size_t)NB * LQ * CM * 2);
    _Float16* vm16  = (_Float16*)alloc((size_t)NB * LQ * CM * 2);
    _Float16* ws16  = (_Float16*)alloc((size_t)NB * LQ * LQ * 2);
    _Float16* fT16  = (_Float16*)alloc((size_t)NB * LQ * CM * 2);
    float*    wmat  = (float*)   alloc((size_t)NB * LQ * LQ * 4);
    _Float16* ff16  = (_Float16*)alloc((size_t)NB * 256 * LQ * 2);

    float* out = (float*)d_out;

    // zero padded buffer (halo) then pack input: NCHW f32 -> padded NHWC f16
    long xpelems = (long)NB * PH * PW * CIN;
    qco_zero16<<<(int)((xpelems + 255) / 256), 256, 0, stream>>>(x16p, xpelems);
    qco_pack_x<<<(NB*CIN*HW + 255)/256, 256, 0, stream>>>((const float*)d_in[0], x16p, (long)NB*CIN*HW);

    for (int br = 0; br < 2; ++br) {
        int b0 = br ? 10 : 1;
        const float* conv1 = (const float*)d_in[b0 + 0];
        const float* bn_g  = (const float*)d_in[b0 + 1];
        const float* bn_b  = (const float*)d_in[b0 + 2];
        const float* conv2 = (const float*)d_in[b0 + 3];
        const float* f1w   = (const float*)d_in[b0 + 4];
        const float* f2w   = (const float*)d_in[b0 + 5];
        const float* outw  = (const float*)d_in[b0 + 6];
        const float* outg  = (const float*)d_in[b0 + 7];
        const float* outb  = (const float*)d_in[b0 + 8];
        const float* kw    = (const float*)d_in[br ? 22 : 19];
        const float* qw    = (const float*)d_in[br ? 23 : 20];
        const float* vw    = (const float*)d_in[br ? 24 : 21];
        const float* aow   = (const float*)d_in[br ? 28 : 25];
        const float* aog   = (const float*)d_in[br ? 29 : 26];
        const float* aob   = (const float*)d_in[br ? 30 : 27];

        // weight packing
        qco_pack_w3<<<(CIN*CIN*9 + 255)/256, 256, 0, stream>>>(conv1, w3p, CIN*CIN*9);
        qco_cvt16<<<(CM*CIN + 255)/256, 256, 0, stream>>>(conv2, w1_16, CM*CIN);
        qco_cvt16<<<(CM*CM + 255)/256, 256, 0, stream>>>(kw, kw16, CM*CM);
        qco_cvt16<<<(CM*CM + 255)/256, 256, 0, stream>>>(qw, qw16, CM*CM);
        qco_cvt16<<<(CM*CM + 255)/256, 256, 0, stream>>>(vw, vw16, CM*CM);
        qco_cvt16<<<(256*CM + 255)/256, 256, 0, stream>>>(aow, ow16, 256*CM);

        // stem: conv3x3 + BN + LeakyReLU -> h16 (NHWC f16); wave = 16 oc x 64 px
        qco_conv3_wmma<<<(16 * (NB*HW/64)) / 8, 256, 0, stream>>>(x16p, w3p, bn_g, bn_b, h16);
        // conv1x1 (256->128) -> xbr (NHWC f32, transposed store)
        launch_gemm(stream, w1_16, CIN, h16, CIN, xbr, CM, 1, nullptr, 0, 0,
                    nullptr, nullptr, 1.f, 0, CM, NB*HW, CIN);

        if (br == 0) {
            qco_reduce_xav<<<NB*CM, 256, 0, stream>>>(xbr, xav, 0);
            qco_xavn<<<NB, 128, 0, stream>>>(xav, xavn);
            qco_cs_c<<<(NB*HW + 255)/256, 256, 0, stream>>>(xbr, xavn, cs);
        } else {
            qco_xsum<<<(NB*HW + 255)/256, 256, 0, stream>>>(xbr, xsum);
            qco_reduce_xav<<<NB*CM, 256, 0, stream>>>(xbr, xav, 1);
            qco_pavg<<<(NB*PP + 255)/256, 256, 0, stream>>>(xsum, pavg);
            qco_normj<<<NB*81, 256, 0, stream>>>(xsum, normj);
            qco_navg<<<NB, 256, 0, stream>>>(pavg, navg);
            qco_folded<<<(NB*HW + 255)/256, 256, 0, stream>>>(xsum, pavg, normj, navg, folded);
            qco_cs_s<<<(NB*HW + 255)/256, 256, 0, stream>>>(folded, cs);
        }

        // soft histogram + quantization
        qco_minmax<<<NB, 256, 0, stream>>>(cs, cmm);
        qco_ql<<<(NB*LQ + 255)/256, 256, 0, stream>>>(cmm, qlv);
        qco_zero<<<(NB*LQ + 255)/256, 256, 0, stream>>>(hist, NB*LQ);
        qco_quant_hist<<<(NB*HW + 255)/256, 256, 0, stream>>>(cs, cmm, quant16, hist);
        qco_hist_norm<<<NB, 128, 0, stream>>>(hist);

        // sta head
        qco_sta1<<<(NB*64*LQ + 255)/256, 256, 0, stream>>>(qlv, hist, f1w, s1);
        qco_sta2<<<(NB*CM*LQ + 255)/256, 256, 0, stream>>>(f2w, s1, s2);
        qco_sta3<<<(NB*CM*LQ + 255)/256, 256, 0, stream>>>(outw, outg, outb, s2, xav, sta16);

        // attention projections (batch folded into Nc=512 columns)
        launch_gemm(stream, kw16, CM, sta16, CM, nullptr, 0, 0, kT16, CM, 1,
                    nullptr, nullptr, 1.f, 0, CM, NB*LQ, CM);
        launch_gemm(stream, qw16, CM, sta16, CM, nullptr, 0, 0, qT16, CM, 1,
                    nullptr, nullptr, 1.f, 0, CM, NB*LQ, CM);
        launch_gemm(stream, vw16, CM, sta16, CM, nullptr, 0, 0, vm16, NB*LQ, 0,
                    nullptr, nullptr, 1.f, 0, CM, NB*LQ, CM);
        // w[l,m] = sum_c k[c,l] q[c,m]
        for (int n = 0; n < NB; ++n)
            launch_gemm(stream, kT16 + (size_t)n*LQ*CM, CM, qT16 + (size_t)n*LQ*CM, CM,
                        wmat + (size_t)n*LQ*LQ, LQ, 0, nullptr, 0, 0,
                        nullptr, nullptr, 1.f, 0, LQ, LQ, CM);
        qco_softmax<<<(NB*LQ + 255)/256, 256, 0, stream>>>(wmat, ws16);
        // f[c,l] = sum_m v[c,m] w[l,m]  -> stored transposed [l][c]
        for (int n = 0; n < NB; ++n)
            launch_gemm(stream, vm16 + (size_t)n*LQ, NB*LQ, ws16 + (size_t)n*LQ*LQ, LQ,
                        nullptr, 0, 0, fT16 + (size_t)n*LQ*CM, CM, 1,
                        nullptr, nullptr, 1.f, 0, CM, LQ, LQ);
        // ff[o,l] = relu(bn(ow @ f)) -> row-major f16 [256][128]
        for (int n = 0; n < NB; ++n)
            launch_gemm(stream, ow16, CM, fT16 + (size_t)n*LQ*CM, CM,
                        nullptr, 0, 0, ff16 + (size_t)n*256*LQ, LQ, 0,
                        aog, aob, BN_SMUL, 1, 256, LQ, CM);
        // o[c,p] = sum_l ff[c,l] quant[p,l] -> d_out rows [n, br*256 + c, p]
        for (int n = 0; n < NB; ++n)
            launch_gemm(stream, ff16 + (size_t)n*256*LQ, LQ, quant16 + (size_t)n*HW*LQ, LQ,
                        out + ((size_t)n*512 + (size_t)br*256) * HW, HW, 0, nullptr, 0, 0,
                        nullptr, nullptr, 1.f, 0, 256, HW, LQ);
    }
}
